// NetworkAction_7645041786903
// MI455X (gfx1250) — matrix operand order repaired and checked by reading, not compiled, not run on hardware
//
#include <hip/hip_runtime.h>

typedef _Float16 v16h __attribute__((ext_vector_type(16)));
typedef _Float16 v8h  __attribute__((ext_vector_type(8)));
typedef float    v8f  __attribute__((ext_vector_type(8)));

#define WMMA_F16(a, b, c) \
  __builtin_amdgcn_wmma_f32_16x16x32_f16(false, (a), false, (b), (short)0, (c), false, false)

static __device__ __forceinline__ void wait_ds0() {
  asm volatile("s_wait_dscnt 0" ::: "memory");
}

// ---------------------------------------------------------------------------
// Kernel 1: fused pairwise network.  One block (4 waves) per agent i.
// Layer 1 computed TRANSPOSED (M=c_out, N=j) so its D-tiles land exactly in
// the register layout the layer-2 A-operand needs: no LDS transpose at all.
//   x(12x16, bias ch) -> [WMMA x4] h1T(64x16) -> relu+pk-cvt (in regs) ->
//   [WMMA x16] h2(16x128) -> fused mask/relu/max-pool.
// Writes pooled (N,128) f32 to workspace.
// ---------------------------------------------------------------------------
__global__ __launch_bounds__(128) void pair_net_kernel(
    const float* __restrict__ s,
    const float* __restrict__ W1, const float* __restrict__ b1,
    const float* __restrict__ W2, const float* __restrict__ b2,
    float* __restrict__ pooled_out) {
  const int i    = blockIdx.x;
  const int tid  = threadIdx.x;
  const int lane = tid & 31;
  const int wave = tid >> 5;
  const int col  = lane & 15;   // j row (layer1 N, layer2 M) / N col of tiles
  const int h    = lane >> 4;   // lane half

  __shared__ float pool[4][128];

  // s_i in registers (uniform)
  float si[10];
#pragma unroll
  for (int c = 0; c < 10; ++c) si[c] = s[i * 10 + c];

  // --- Layer 1 weights as A operand (M=c_out, K=c_in; K=11 carries bias) ---
  // A layout: elem e -> K = 16h + e, row M = col (within tile).
  v16h Aw[4];
#pragma unroll
  for (int t = 0; t < 4; ++t) {
#pragma unroll
    for (int e = 0; e < 16; ++e) {
      const int K = 16 * h + e;
      float w = 0.0f;
      if (K < 11)       w = W1[(16 * t + col) * 11 + K];
      else if (K == 11) w = b1[16 * t + col];
      Aw[t][e] = (_Float16)w;
    }
  }

  // --- Layer 2 B operands (W2: 128x64) -> 8 N-tiles x 2 K-steps ---
  v16h B2[8][2];
#pragma unroll
  for (int t = 0; t < 8; ++t) {
#pragma unroll
    for (int ks = 0; ks < 2; ++ks) {
#pragma unroll
      for (int e = 0; e < 16; ++e) {
        B2[t][ks][e] =
            (_Float16)W2[(16 * t + col) * 64 + 32 * ks + 8 * h + 16 * (e >> 3) + (e & 7)];
      }
    }
  }
  // Layer-2 bias pre-splatted as C operands (D != C, so no per-iter movs)
  v8f cb2[8];
#pragma unroll
  for (int t = 0; t < 8; ++t) {
    const float bv = b2[16 * t + col];
#pragma unroll
    for (int r = 0; r < 8; ++r) cb2[t][r] = bv;
  }

  v8f zc;
#pragma unroll
  for (int r = 0; r < 8; ++r) zc[r] = 0.0f;

  float p[8];
#pragma unroll
  for (int t = 0; t < 8; ++t) p[t] = 0.0f;

  const bool lowhalf = (h == 0);

  for (int jt = wave; jt < 64; jt += 4) {
    const int jj = jt * 16 + col;

    // s_j: 5x float2 (rows are 40B, 8B-aligned)
    const float2* sj = (const float2*)(s + jj * 10);
    const float2 q0 = sj[0], q1 = sj[1], q2 = sj[2], q3 = sj[3], q4 = sj[4];
    const float d0 = si[0] - q0.x, d1 = si[1] - q0.y;
    const float d2 = si[2] - q1.x, d3 = si[3] - q1.y;
    const float d4 = si[4] - q2.x, d5 = si[5] - q2.y;
    const float d6 = si[6] - q3.x, d7 = si[7] - q3.y;
    const float d8 = si[8] - q4.x, d9 = si[9] - q4.y;
    const float eye = (jj == i) ? 1.0f : 0.0f;
    const float dist2 = d0 * d0 + d1 * d1 + d2 * d2;
    const unsigned mask = __builtin_amdgcn_ballot_w32(dist2 < 4.0f);
    const unsigned mshift = mask >> (8 * h);
    float mf[8];
#pragma unroll
    for (int r = 0; r < 8; ++r)
      mf[r] = (mshift & (1u << r)) ? 1.0f : 0.0f;

    // B operand for layer 1: lane half 0 carries K=0..15 (x, eye, bias=1),
    // lane half 1 carries K=16..31 = 0.  Constant indices -> plain selects.
    v16h Bx;
    Bx[0]  = lowhalf ? (_Float16)d0 : (_Float16)0.0f;
    Bx[1]  = lowhalf ? (_Float16)d1 : (_Float16)0.0f;
    Bx[2]  = lowhalf ? (_Float16)d2 : (_Float16)0.0f;
    Bx[3]  = lowhalf ? (_Float16)d3 : (_Float16)0.0f;
    Bx[4]  = lowhalf ? (_Float16)d4 : (_Float16)0.0f;
    Bx[5]  = lowhalf ? (_Float16)d5 : (_Float16)0.0f;
    Bx[6]  = lowhalf ? (_Float16)d6 : (_Float16)0.0f;
    Bx[7]  = lowhalf ? (_Float16)d7 : (_Float16)0.0f;
    Bx[8]  = lowhalf ? (_Float16)d8 : (_Float16)0.0f;
    Bx[9]  = lowhalf ? (_Float16)d9 : (_Float16)0.0f;
    Bx[10] = lowhalf ? (_Float16)eye : (_Float16)0.0f;
    Bx[11] = lowhalf ? (_Float16)1.0f : (_Float16)0.0f;  // bias channel
    Bx[12] = (_Float16)0.0f;
    Bx[13] = (_Float16)0.0f;
    Bx[14] = (_Float16)0.0f;
    Bx[15] = (_Float16)0.0f;

    // Layer 1 (transposed): 4 tiles of h1T, C = 0
    v8f c1[4];
#pragma unroll
    for (int t = 0; t < 4; ++t) c1[t] = WMMA_F16(Aw[t], Bx, zc);

    // relu + pack straight into the layer-2 A operand (pure register work):
    // A2[st] elems 0..7 = tile 2st vgprs 0..7, elems 8..15 = tile 2st+1.
    v16h A2[2];
#pragma unroll
    for (int st = 0; st < 2; ++st) {
#pragma unroll
      for (int e = 0; e < 8; ++e) {
        A2[st][e]     = (_Float16)fmaxf(c1[2 * st][e], 0.0f);
        A2[st][e + 8] = (_Float16)fmaxf(c1[2 * st + 1][e], 0.0f);
      }
    }

    // Layer 2 + fused mask/relu/max-pool over j (=M rows)
#pragma unroll
    for (int t = 0; t < 8; ++t) {
      v8f c = WMMA_F16(A2[0], B2[t][0], cb2[t]);
      c = WMMA_F16(A2[1], B2[t][1], c);
#pragma unroll
      for (int r = 0; r < 8; ++r)
        p[t] = fmaxf(p[t], c[r] * mf[r]);  // p>=0 makes this == masked relu-max
    }
  }

  // Combine lane halves (same N, different M groups), then cross-wave reduce
#pragma unroll
  for (int t = 0; t < 8; ++t) {
    const float o = fmaxf(p[t], __shfl_xor(p[t], 16, 32));
    if (h == 0) pool[wave][16 * t + col] = o;
  }
  __syncthreads();
  {
    const int ch = tid;  // 0..127
    const float m = fmaxf(fmaxf(pool[0][ch], pool[1][ch]),
                          fmaxf(pool[2][ch], pool[3][ch]));
    pooled_out[i * 128 + ch] = m;
  }
}

// ---------------------------------------------------------------------------
// Kernel 2: MLP head, 16 rows per block, single wave, all layers in WMMA.
// feat = [pooled(128), s - s_ref(10)] (K padded to 160) -> 64 -> 128 -> 64 -> 10
// out = y + (s_ref - s) / 0.1
// ---------------------------------------------------------------------------
__global__ __launch_bounds__(32) void head_kernel(
    const float* __restrict__ pooled,
    const float* __restrict__ s, const float* __restrict__ sref,
    const float* __restrict__ Wf1, const float* __restrict__ bf1,
    const float* __restrict__ Wf2, const float* __restrict__ bf2,
    const float* __restrict__ Wf3, const float* __restrict__ bf3,
    const float* __restrict__ Wf4, const float* __restrict__ bf4,
    float* __restrict__ out) {
  const int m0   = blockIdx.x * 16;
  const int lane = threadIdx.x & 31;
  const int col  = lane & 15;
  const int h    = lane >> 4;
  const int row  = m0 + col;

  __shared__ alignas(16) _Float16 bufA[16 * 64];
  __shared__ alignas(16) _Float16 bufB[16 * 128];

  // A operand: feature vector, 5 k-steps of 32
  v16h Af[5];
#pragma unroll
  for (int st = 0; st < 5; ++st) {
#pragma unroll
    for (int e = 0; e < 16; ++e) {
      const int K = 32 * st + 8 * h + 16 * (e >> 3) + (e & 7);
      float v = 0.0f;
      if (K < 128)      v = pooled[row * 128 + K];
      else if (K < 138) v = s[row * 10 + (K - 128)] - sref[row * 10 + (K - 128)];
      Af[st][e] = (_Float16)v;
    }
  }

  // f1: (138 -> 64)
#pragma unroll
  for (int t = 0; t < 4; ++t) {
    v8f c;
    const float bias = bf1[16 * t + col];
#pragma unroll
    for (int r = 0; r < 8; ++r) c[r] = bias;
#pragma unroll
    for (int st = 0; st < 5; ++st) {
      v16h B;
#pragma unroll
      for (int e = 0; e < 16; ++e) {
        const int K = 32 * st + 8 * h + 16 * (e >> 3) + (e & 7);
        B[e] = (_Float16)((K < 138) ? Wf1[(16 * t + col) * 138 + K] : 0.0f);
      }
      c = WMMA_F16(Af[st], B, c);
    }
#pragma unroll
    for (int r = 0; r < 8; ++r)
      bufA[(r + 8 * h) * 64 + 16 * t + col] = (_Float16)fmaxf(c[r], 0.0f);
  }
  wait_ds0();

  // f2: (64 -> 128)
  v16h A2[2];
#pragma unroll
  for (int st = 0; st < 2; ++st) {
    const v8h lo = *(const v8h*)(bufA + col * 64 + 32 * st + 8 * h);
    const v8h hi = *(const v8h*)(bufA + col * 64 + 32 * st + 8 * h + 16);
#pragma unroll
    for (int e = 0; e < 8; ++e) { A2[st][e] = lo[e]; A2[st][e + 8] = hi[e]; }
  }
#pragma unroll
  for (int t = 0; t < 8; ++t) {
    v8f c;
    const float bias = bf2[16 * t + col];
#pragma unroll
    for (int r = 0; r < 8; ++r) c[r] = bias;
#pragma unroll
    for (int st = 0; st < 2; ++st) {
      v16h B;
#pragma unroll
      for (int e = 0; e < 16; ++e)
        B[e] = (_Float16)Wf2[(16 * t + col) * 64 + 32 * st + 8 * h + 16 * (e >> 3) + (e & 7)];
      c = WMMA_F16(A2[st], B, c);
    }
#pragma unroll
    for (int r = 0; r < 8; ++r)
      bufB[(r + 8 * h) * 128 + 16 * t + col] = (_Float16)fmaxf(c[r], 0.0f);
  }
  wait_ds0();

  // f3: (128 -> 64)
  v16h A3[4];
#pragma unroll
  for (int st = 0; st < 4; ++st) {
    const v8h lo = *(const v8h*)(bufB + col * 128 + 32 * st + 8 * h);
    const v8h hi = *(const v8h*)(bufB + col * 128 + 32 * st + 8 * h + 16);
#pragma unroll
    for (int e = 0; e < 8; ++e) { A3[st][e] = lo[e]; A3[st][e + 8] = hi[e]; }
  }
#pragma unroll
  for (int t = 0; t < 4; ++t) {
    v8f c;
    const float bias = bf3[16 * t + col];
#pragma unroll
    for (int r = 0; r < 8; ++r) c[r] = bias;
#pragma unroll
    for (int st = 0; st < 4; ++st) {
      v16h B;
#pragma unroll
      for (int e = 0; e < 16; ++e)
        B[e] = (_Float16)Wf3[(16 * t + col) * 128 + 32 * st + 8 * h + 16 * (e >> 3) + (e & 7)];
      c = WMMA_F16(A3[st], B, c);
    }
#pragma unroll
    for (int r = 0; r < 8; ++r)
      bufA[(r + 8 * h) * 64 + 16 * t + col] = (_Float16)fmaxf(c[r], 0.0f);
  }
  wait_ds0();

  // f4: (64 -> 10), fused epilogue out = y + (s_ref - s) * 10
  v16h A4[2];
#pragma unroll
  for (int st = 0; st < 2; ++st) {
    const v8h lo = *(const v8h*)(bufA + col * 64 + 32 * st + 8 * h);
    const v8h hi = *(const v8h*)(bufA + col * 64 + 32 * st + 8 * h + 16);
#pragma unroll
    for (int e = 0; e < 8; ++e) { A4[st][e] = lo[e]; A4[st][e + 8] = hi[e]; }
  }
  {
    v8f c;
    const float bias = (col < 10) ? bf4[col] : 0.0f;
#pragma unroll
    for (int r = 0; r < 8; ++r) c[r] = bias;
#pragma unroll
    for (int st = 0; st < 2; ++st) {
      v16h B;
#pragma unroll
      for (int e = 0; e < 16; ++e) {
        const int K = 32 * st + 8 * h + 16 * (e >> 3) + (e & 7);
        B[e] = (_Float16)((col < 10) ? Wf4[col * 64 + K] : 0.0f);
      }
      c = WMMA_F16(A4[st], B, c);
    }
    if (col < 10) {
#pragma unroll
      for (int r = 0; r < 8; ++r) {
        const int rr = m0 + r + 8 * h;
        out[rr * 10 + col] =
            c[r] + (sref[rr * 10 + col] - s[rr * 10 + col]) * 10.0f;
      }
    }
  }
}

extern "C" void kernel_launch(void* const* d_in, const int* in_sizes, int n_in,
                              void* d_out, int out_size, void* d_ws,
                              size_t ws_size, hipStream_t stream) {
  const float* s    = (const float*)d_in[0];
  const float* sref = (const float*)d_in[1];
  const float* W1   = (const float*)d_in[2];
  const float* b1   = (const float*)d_in[3];
  const float* W2   = (const float*)d_in[4];
  const float* b2   = (const float*)d_in[5];
  const float* Wf1  = (const float*)d_in[6];
  const float* bf1  = (const float*)d_in[7];
  const float* Wf2  = (const float*)d_in[8];
  const float* bf2  = (const float*)d_in[9];
  const float* Wf3  = (const float*)d_in[10];
  const float* bf3  = (const float*)d_in[11];
  const float* Wf4  = (const float*)d_in[12];
  const float* bf4  = (const float*)d_in[13];

  float* pooled = (float*)d_ws;  // 1024 * 128 f32 = 512 KB

  pair_net_kernel<<<1024, 128, 0, stream>>>(s, W1, b1, W2, b2, pooled);
  head_kernel<<<64, 32, 0, stream>>>(pooled, s, sref, Wf1, bf1, Wf2, bf2, Wf3,
                                     bf3, Wf4, bf4, (float*)d_out);
}
